// Heterogeneous_GNN_38182259261838
// MI455X (gfx1250) — matrix operand order repaired and checked by reading, not compiled, not run on hardware
//
#include <hip/hip_runtime.h>
#include <hip/hip_bf16.h>
#include <math.h>

// ---------------------------------------------------------------------------
// MI455X (gfx1250) heterogeneous-GAT forward.
//  - All GEMM operands pre-converted to f16 (A row-major, B transposed N x K),
//    so the WMMA GEMM stages LDS tiles with raw 16B vector copies.
//  - 128x128x32 block tile, 8 waves, 8 x v_wmma_f32_16x16x32_f16 per wave/step,
//    register-staged software pipeline for global->LDS; all B fragments
//    preloaded so the 8 WMMAs issue back-to-back after one dscnt drain.
//  - Edge-feature attention collapsed to ef @ (We . attn_e)  (768->8).
//  - fs/fd aliased when src_type == dst_type (rel, rel_inv).
//  - Edge softmax: atomic segment-max (sign-split int trick) + atomic segsum.
// ---------------------------------------------------------------------------

typedef _Float16 half_t;
typedef __attribute__((ext_vector_type(16))) _Float16 v16h;
typedef __attribute__((ext_vector_type(8)))  float    v8f;

#define D_MODEL 768
#define N_HEADS 8
#define D_HEAD  96

// ------------------------------ WMMA GEMM ----------------------------------
// C[M,N] = A[M,K] @ Bt[N,K]^T (+bias) (+relu). A,Bt f16; C f32.
#define BM 128
#define BN 128
#define BK 32
#define LDT 40   // halfword stride (32 + 8 pad), keeps 16B alignment (80B rows)

__device__ __forceinline__ uint4 load8h(const half_t* __restrict__ p, long row,
                                        int LD, int kk, int RMAX, bool fast) {
  if (fast) return *(const uint4*)(p + row * (long)LD + kk);
  union { uint4 v; half_t h[8]; } u;
  #pragma unroll
  for (int j = 0; j < 8; ++j) {
    bool ok = (row < (long)RMAX) && (kk + j < LD);
    u.h[j] = ok ? p[row * (long)LD + kk + j] : (half_t)0.0f;
  }
  return u.v;
}

__global__ __launch_bounds__(256) void k_gemm_wmma(
    const half_t* __restrict__ A, const half_t* __restrict__ Bt,
    const float* __restrict__ bias, float* __restrict__ C,
    int M, int N, int K, int use_bias, int do_relu)
{
  __shared__ half_t As[BM * LDT];
  __shared__ half_t Bs[BN * LDT];

  const int tid  = threadIdx.x;
  const int wave = tid >> 5;
  const int lane = tid & 31;
  const int lm   = lane & 15;
  const int hi   = lane >> 4;            // 0 or 1
  const int bm   = blockIdx.x * BM;
  const int bn   = blockIdx.y * BN;

  // staging assignment: thread owns rows (tid>>2) and (tid>>2)+64,
  // 8-half segment at k offset (tid&3)*8  (tile = 128 rows x 32 halfs)
  const int rS = tid >> 2;
  const int kS = (tid & 3) * 8;

  v8f acc[8];
  #pragma unroll
  for (int c = 0; c < 8; ++c)
    #pragma unroll
    for (int j = 0; j < 8; ++j) acc[c][j] = 0.0f;

  const int ktiles = (K + BK - 1) / BK;
  uint4 aR0, aR1, bR0, bR1;

  auto stage = [&](int kt) {
    const int k0 = kt * BK;
    const bool fA = (bm + BM <= M) && (k0 + BK <= K);
    const bool fB = (bn + BN <= N) && (k0 + BK <= K);
    aR0 = load8h(A,  (long)(bm + rS),      K, k0 + kS, M, fA);
    aR1 = load8h(A,  (long)(bm + rS + 64), K, k0 + kS, M, fA);
    bR0 = load8h(Bt, (long)(bn + rS),      K, k0 + kS, N, fB);
    bR1 = load8h(Bt, (long)(bn + rS + 64), K, k0 + kS, N, fB);
  };
  auto commit = [&]() {
    *(uint4*)&As[rS * LDT + kS]        = aR0;
    *(uint4*)&As[(rS + 64) * LDT + kS] = aR1;
    *(uint4*)&Bs[rS * LDT + kS]        = bR0;
    *(uint4*)&Bs[(rS + 64) * LDT + kS] = bR1;
  };

  stage(0);
  commit();
  __syncthreads();

  for (int kt = 0;;) {
    const bool more = (kt + 1 < ktiles);
    if (more) stage(kt + 1);              // overlap global loads with WMMA

    // A fragment (16x32 f16): lane m = lm, kbase = hi*8;
    // elems 0..7 -> K=kb..kb+7, elems 8..15 -> K=kb+16..kb+23
    const half_t* arow = &As[(wave * 16 + lm) * LDT];
    union { uint4 q[2]; v16h v; } af;
    const int kb = hi * 8;
    af.q[0] = *(const uint4*)(arow + kb);
    af.q[1] = *(const uint4*)(arow + kb + 16);

    // preload ALL B fragments, then issue the 8 WMMAs back-to-back
    union BF { uint4 q[2]; v16h v; } bf[8];
    #pragma unroll
    for (int cg = 0; cg < 8; ++cg) {
      // B fragment (32x16 f16): lane n = lm, elems 0..7 -> K = hi*8 + i, elems 8..15 -> K = 16 + hi*8 + (i-8)
      const half_t* brow = &Bs[(cg * 16 + lm) * LDT + hi * 8];
      bf[cg].q[0] = *(const uint4*)(brow);
      bf[cg].q[1] = *(const uint4*)(brow + 16);
    }
    #pragma unroll
    for (int cg = 0; cg < 8; ++cg) {
      acc[cg] = __builtin_amdgcn_wmma_f32_16x16x32_f16(
          false, af.v, false, bf[cg].v, (short)0, acc[cg], false, false);
    }

    if (!more) break;
    __syncthreads();
    commit();
    __syncthreads();
    ++kt;
  }

  // store: elem r -> row = wave*16 + r + hi*8; col = cg*16 + lm
  #pragma unroll
  for (int cg = 0; cg < 8; ++cg) {
    int gn = bn + cg * 16 + lm;
    if (gn >= N) continue;
    float bv = use_bias ? bias[gn] : 0.0f;
    #pragma unroll
    for (int r = 0; r < 8; ++r) {
      int gm = bm + wave * 16 + r + hi * 8;
      if (gm < M) {
        float v = acc[cg][r] + bv;
        if (do_relu) v = v > 0.f ? v : 0.f;
        C[(long)gm * N + gn] = v;
      }
    }
  }
}

// --------------------------- conversion kernels ----------------------------
__global__ void k_cvt(half_t* o, const float* x, long n) {   // n multiple of 4
  long i4 = ((long)blockIdx.x * blockDim.x + threadIdx.x) * 4;
  if (i4 + 3 < n) {
    float4 v = *(const float4*)(x + i4);
    union { uint2 u; half_t h[4]; } w;
    w.h[0] = (half_t)v.x; w.h[1] = (half_t)v.y;
    w.h[2] = (half_t)v.z; w.h[3] = (half_t)v.w;
    *(uint2*)(o + i4) = w.u;
  } else {
    for (long i = i4; i < n; ++i) o[i] = (half_t)x[i];
  }
}
__global__ void k_relu_cvt(half_t* o, const float* x, long n) {
  long i4 = ((long)blockIdx.x * blockDim.x + threadIdx.x) * 4;
  if (i4 + 3 < n) {
    float4 v = *(const float4*)(x + i4);
    union { uint2 u; half_t h[4]; } w;
    w.h[0] = (half_t)fmaxf(v.x, 0.f); w.h[1] = (half_t)fmaxf(v.y, 0.f);
    w.h[2] = (half_t)fmaxf(v.z, 0.f); w.h[3] = (half_t)fmaxf(v.w, 0.f);
    *(uint2*)(o + i4) = w.u;
  } else {
    for (long i = i4; i < n; ++i) o[i] = (half_t)fmaxf(x[i], 0.f);
  }
}
// Bt[n,k] = (f16) W[k,n]     (W: K x N row-major)
__global__ void k_cvtT(half_t* o, const float* w, int K, int N) {
  long i = (long)blockIdx.x * blockDim.x + threadIdx.x;
  if (i >= (long)K * N) return;
  int n = (int)(i / K), k = (int)(i % K);
  o[i] = (half_t)w[(long)k * N + n];
}

// ------------------------------ small kernels ------------------------------
__global__ void k_zero(float* p, long n) {
  long i = (long)blockIdx.x * blockDim.x + threadIdx.x;
  if (i < n) p[i] = 0.0f;
}
__global__ void k_fill(float* p, long n, float v) {
  long i = (long)blockIdx.x * blockDim.x + threadIdx.x;
  if (i < n) p[i] = v;
}
__global__ void k_i2f(float* o, const int* x, int n) {
  int i = blockIdx.x * blockDim.x + threadIdx.x;
  if (i < n) o[i] = (float)x[i];
}

// el[n,h] = sum_d feat[n, h*96+d] * attn[h,d]
__global__ void k_headdot(float* out, const float* feat, const float* attn, int n) {
  int i = blockIdx.x * blockDim.x + threadIdx.x;
  if (i >= n * N_HEADS) return;
  int node = i >> 3, h = i & 7;
  const float* f = feat + (long)node * D_MODEL + h * D_HEAD;
  const float* a = attn + h * D_HEAD;
  float s = 0.f;
  #pragma unroll
  for (int d = 0; d < D_HEAD; d += 4) {
    float4 fv = *(const float4*)(f + d);
    float4 av = *(const float4*)(a + d);
    s += fv.x * av.x + fv.y * av.y + fv.z * av.z + fv.w * av.w;
  }
  out[i] = s;
}

// wev[k,h] = sum_d We[k, h*96+d] * attn_e[h,d]   (768x8)
__global__ void k_wevec(float* wev, const float* We, const float* attn_e) {
  int i = blockIdx.x * blockDim.x + threadIdx.x;
  if (i >= D_MODEL * N_HEADS) return;
  int k = i >> 3, h = i & 7;
  const float* w = We + (long)k * D_MODEL + h * D_HEAD;
  const float* a = attn_e + h * D_HEAD;
  float s = 0.f;
  #pragma unroll
  for (int d = 0; d < D_HEAD; d += 4) {
    float4 wv = *(const float4*)(w + d);
    float4 av = *(const float4*)(a + d);
    s += wv.x * av.x + wv.y * av.y + wv.z * av.z + wv.w * av.w;
  }
  wev[k * N_HEADS + h] = s;
}

// ele[e,h] = sum_k ef[e,k] * wev[k,h]
__global__ void k_edgefeat(float* ele, const float* ef, const float* wev, int E) {
  int i = blockIdx.x * blockDim.x + threadIdx.x;
  if (i >= E * N_HEADS) return;
  int e = i >> 3, h = i & 7;
  const float* x = ef + (long)e * D_MODEL;
  float s = 0.f;
  for (int k = 0; k < D_MODEL; k += 4) {
    float4 xv = *(const float4*)(x + k);
    s += xv.x * wev[(k + 0) * N_HEADS + h] + xv.y * wev[(k + 1) * N_HEADS + h]
       + xv.z * wev[(k + 2) * N_HEADS + h] + xv.w * wev[(k + 3) * N_HEADS + h];
  }
  ele[i] = s;
}

__global__ void k_logit(float* logit, const float* el, const float* er,
                        const float* ele, const int* src, const int* dst, int E) {
  int i = blockIdx.x * blockDim.x + threadIdx.x;
  if (i >= E * N_HEADS) return;
  int e = i >> 3, h = i & 7;
  float v = el[src[e] * N_HEADS + h] + er[dst[e] * N_HEADS + h];
  if (ele) v += ele[i];
  logit[i] = v > 0.f ? v : 0.2f * v;      // leaky_relu 0.2
}

__device__ __forceinline__ void atomicMaxF(float* a, float v) {
  if (v >= 0.f) atomicMax((int*)a, __float_as_int(v));
  else          atomicMin((unsigned int*)a, __float_as_uint(v));
}

__global__ void k_segmax(float* m, const float* logit, const int* dst, int E) {
  int i = blockIdx.x * blockDim.x + threadIdx.x;
  if (i >= E * N_HEADS) return;
  int e = i >> 3, h = i & 7;
  atomicMaxF(&m[dst[e] * N_HEADS + h], logit[i]);
}

__global__ void k_expsum(float* logit, const float* m, float* s, const int* dst, int E) {
  int i = blockIdx.x * blockDim.x + threadIdx.x;
  if (i >= E * N_HEADS) return;
  int e = i >> 3, h = i & 7;
  float ev = expf(logit[i] - m[dst[e] * N_HEADS + h]);
  logit[i] = ev;
  atomicAdd(&s[dst[e] * N_HEADS + h], ev);
}

__global__ void k_alpha(float* logit, const float* s, const int* dst, int E) {
  int i = blockIdx.x * blockDim.x + threadIdx.x;
  if (i >= E * N_HEADS) return;
  int e = i >> 3, h = i & 7;
  logit[i] /= (s[dst[e] * N_HEADS + h] + 1e-9f);
}

// gat[dst, d] += alpha[e, d/96] * fs[src, d]
__global__ void k_agg(float* gat, const float* alpha, const float* fs,
                      const int* src, const int* dst, int E) {
  long i = (long)blockIdx.x * blockDim.x + threadIdx.x;
  long tot = (long)E * (D_MODEL / 4);
  if (i >= tot) return;
  int e = (int)(i / (D_MODEL / 4));
  int d = (int)(i % (D_MODEL / 4)) * 4;
  int h = d / D_HEAD;
  float a = alpha[e * N_HEADS + h];
  float4 f = *(const float4*)(fs + (long)src[e] * D_MODEL + d);
  float* o = gat + (long)dst[e] * D_MODEL + d;
  atomicAdd(&o[0], a * f.x);
  atomicAdd(&o[1], a * f.y);
  atomicAdd(&o[2], a * f.z);
  atomicAdd(&o[3], a * f.w);
}

__global__ void k_addbias(float* hn, const float* gat, const float* bias, long n) {
  long i = (long)blockIdx.x * blockDim.x + threadIdx.x;
  if (i < n) hn[i] += gat[i] + bias[i % D_MODEL];
}
__global__ void k_y(float* y, const float* gat, const float* fd, const float* bias, long n) {
  long i = (long)blockIdx.x * blockDim.x + threadIdx.x;
  if (i < n) y[i] = gat[i] + fd[i] + bias[i % D_MODEL];
}

__global__ void k_bnstats(float* mean, float* var, const float* z, int M) {
  int d = blockIdx.x;                     // one block per column (768)
  __shared__ float ss[256], sq[256];
  float s = 0.f, q = 0.f;
  for (int r = threadIdx.x; r < M; r += 256) {
    float v = z[(long)r * D_MODEL + d];
    s += v; q += v * v;
  }
  ss[threadIdx.x] = s; sq[threadIdx.x] = q;
  __syncthreads();
  for (int off = 128; off > 0; off >>= 1) {
    if (threadIdx.x < off) { ss[threadIdx.x] += ss[threadIdx.x + off];
                             sq[threadIdx.x] += sq[threadIdx.x + off]; }
    __syncthreads();
  }
  if (threadIdx.x == 0) {
    float mu = ss[0] / (float)M;
    mean[d] = mu;
    var[d]  = sq[0] / (float)M - mu * mu;
  }
}
__global__ void k_bnapply(float* hn, const float* z, const float* mean, const float* var,
                          const float* gamma, const float* beta, long n) {
  long i = (long)blockIdx.x * blockDim.x + threadIdx.x;
  if (i >= n) return;
  int d = (int)(i % D_MODEL);
  hn[i] += gamma[d] * (z[i] - mean[d]) * rsqrtf(var[d] + 1e-5f) + beta[d];
}

__global__ void k_ccount(float* cnt, const int* dst, int E) {
  int i = blockIdx.x * blockDim.x + threadIdx.x;
  if (i < E) atomicAdd(&cnt[dst[i]], 1.0f);
}
__global__ void k_csum(float* xc, const float* xev, const int* src, const int* dst, int E) {
  long i = (long)blockIdx.x * blockDim.x + threadIdx.x;
  long tot = (long)E * (D_MODEL / 4);
  if (i >= tot) return;
  int e = (int)(i / (D_MODEL / 4));
  int d = (int)(i % (D_MODEL / 4)) * 4;
  float4 f = *(const float4*)(xev + (long)src[e] * D_MODEL + d);
  float* o = xc + (long)dst[e] * D_MODEL + d;
  atomicAdd(&o[0], f.x); atomicAdd(&o[1], f.y);
  atomicAdd(&o[2], f.z); atomicAdd(&o[3], f.w);
}
__global__ void k_cdiv(float* xc, const float* cnt, long n) {
  long i = (long)blockIdx.x * blockDim.x + threadIdx.x;
  if (i < n) xc[i] /= fmaxf(cnt[i / D_MODEL], 1.0f);
}

// ------------------------------ host driver --------------------------------
extern "C" void kernel_launch(void* const* d_in, const int* in_sizes, int n_in,
                              void* d_out, int out_size, void* d_ws, size_t ws_size,
                              hipStream_t stream) {
  (void)in_sizes; (void)n_in; (void)out_size; (void)ws_size;

  auto F = [&](int i) { return (const float*)d_in[i]; };
  auto I = [&](int i) { return (const int*)d_in[i]; };

  // node sizes: doc, ent, event, center
  const int  NS[4]   = {1000, 20000, 8000, 200};
  const long HOFF[4] = {0L, 1000L * D_MODEL, 21000L * D_MODEL, 29000L * D_MODEL};
  const long NT = 29200L * D_MODEL;

  // workspace carve (f32 region then f16 region; all offsets keep 16B align)
  float* ws  = (float*)d_ws;
  float* hA  = ws;  ws += NT;
  float* hB  = ws;  ws += NT;
  float* fs  = ws;  ws += 20000L * D_MODEL;
  float* fd  = ws;  ws += 20000L * D_MODEL;
  float* gat = ws;  ws += 20000L * D_MODEL;
  float* el  = ws;  ws += 160000;
  float* er  = ws;  ws += 160000;
  float* ele = ws;  ws += 480000;
  float* lgt = ws;  ws += 480000;
  float* smx = ws;  ws += 160000;
  float* ssm = ws;  ws += 160000;
  float* wev = ws;  ws += D_MODEL * N_HEADS;
  float* bmu = ws;  ws += D_MODEL;
  float* bvr = ws;  ws += D_MODEL;
  float* cnt = ws;  ws += 200;
  half_t* h16  = (half_t*)ws;          // 29200*768 halfs
  half_t* a16  = h16 + NT;             // 20000*768 halfs (dynamic A operands)
  half_t* w16t = a16 + 20000L * D_MODEL;  // up to 768*768 halfs (weights^T)

  auto g1 = [](long n) { return dim3((unsigned)((n + 255) / 256)); };
  auto gemm = [&](const half_t* A, const half_t* Bt, const float* bias, float* C,
                  int M, int N, int K, int ub, int rl) {
    dim3 grid((M + BM - 1) / BM, (N + BN - 1) / BN);
    k_gemm_wmma<<<grid, 256, 0, stream>>>(A, Bt, bias, C, M, N, K, ub, rl);
  };
  auto cvt  = [&](half_t* o, const float* x, long n) {
    k_cvt<<<g1((n + 3) / 4), 256, 0, stream>>>(o, x, n);
  };
  auto cvtT = [&](half_t* o, const float* w, int K, int N) {
    k_cvtT<<<g1((long)K * N), 256, 0, stream>>>(o, w, K, N);
  };

  // ---- relation table (REL order) ----
  // params per layer: sorted keys. edge: W,We,attn_e,attn_l,attn_r,bias
  //                   wrap: W,W_out,attn_l,attn_r,b_out,beta,bias,gamma
  struct Rel { int st, dt, has_e, si, di, efi, E, po; };
  const Rel rels[10] = {
    {1, 1, 1,  7,  8, 3, 60000, 60},  // rel
    {1, 1, 1,  9, 10, 4, 60000, 66},  // rel_inv
    {2, 1, 1, 11, 12, 5, 40000,  0},  // arg
    {1, 2, 1, 13, 14, 6, 40000,  6},  // arg_inv
    {1, 0, 0, 15, 16,-1, 60000, 36},  // ent2doc
    {0, 1, 0, 17, 18,-1, 60000, 20},  // doc2ent
    {2, 0, 0, 19, 20,-1, 30000, 52},  // event2doc
    {0, 2, 0, 21, 22,-1, 30000, 28},  // doc2event
    {2, 3, 0, 23, 24,-1,  8000, 44},  // event2center
    {3, 2, 0, 25, 26,-1,  8000, 12},  // center2event
  };

  // ---- init h: doc/ent/event copy; center = masked mean of event feats ----
  hipMemcpyAsync(hA + HOFF[0], d_in[0], (size_t)NS[0] * D_MODEL * 4, hipMemcpyDeviceToDevice, stream);
  hipMemcpyAsync(hA + HOFF[1], d_in[1], (size_t)NS[1] * D_MODEL * 4, hipMemcpyDeviceToDevice, stream);
  hipMemcpyAsync(hA + HOFF[2], d_in[2], (size_t)NS[2] * D_MODEL * 4, hipMemcpyDeviceToDevice, stream);
  {
    long nc = (long)NS[3] * D_MODEL;
    k_zero<<<g1(nc), 256, 0, stream>>>(hA + HOFF[3], nc);
    k_zero<<<g1(200), 256, 0, stream>>>(cnt, 200);
    int E = 8000;  // event2center
    k_ccount<<<g1(E), 256, 0, stream>>>(cnt, I(24), E);
    k_csum<<<g1((long)E * 192), 256, 0, stream>>>(hA + HOFF[3], F(2), I(23), I(24), E);
    k_cdiv<<<g1(nc), 256, 0, stream>>>(hA + HOFF[3], cnt, nc);
  }

  float* hbuf[2] = {hA, hB};
  int cur = 0;

  for (int l = 0; l < 2; ++l) {
    const int LB = 41 + l * 72;          // layer param leaf base
    float* hc = hbuf[cur];
    float* hn = hbuf[cur ^ 1];
    k_zero<<<g1(NT), 256, 0, stream>>>(hn, NT);
    cvt(h16, hc, NT);                    // f16 node features, shared by all rels

    for (int r = 0; r < 10; ++r) {
      const Rel& R = rels[r];
      const int Ns = NS[R.st], Nd = NS[R.dt], E = R.E;
      const float* W      = F(LB + R.po + 0);
      const float* attn_l = R.has_e ? F(LB + R.po + 3) : F(LB + R.po + 2);
      const float* attn_r = R.has_e ? F(LB + R.po + 4) : F(LB + R.po + 3);
      const float* bias   = R.has_e ? F(LB + R.po + 5) : F(LB + R.po + 6);

      // projections (WMMA): fs = h[st] @ W ; fd = h[dt] @ W (aliased if st==dt)
      cvtT(w16t, W, D_MODEL, D_MODEL);
      gemm(h16 + HOFF[R.st], w16t, nullptr, fs, Ns, D_MODEL, D_MODEL, 0, 0);
      const float* fdp = fs;
      if (R.st != R.dt) {
        gemm(h16 + HOFF[R.dt], w16t, nullptr, fd, Nd, D_MODEL, D_MODEL, 0, 0);
        fdp = fd;
      }

      // attention scalars
      k_headdot<<<g1((long)Ns * 8), 256, 0, stream>>>(el, fs,  attn_l, Ns);
      k_headdot<<<g1((long)Nd * 8), 256, 0, stream>>>(er, fdp, attn_r, Nd);
      const float* elep = nullptr;
      if (R.has_e) {
        k_wevec<<<g1(D_MODEL * 8), 256, 0, stream>>>(wev, F(LB + R.po + 1), F(LB + R.po + 2));
        k_edgefeat<<<g1((long)E * 8), 256, 0, stream>>>(ele, F(R.efi), wev, E);
        elep = ele;
      }
      k_logit<<<g1((long)E * 8), 256, 0, stream>>>(lgt, el, er, elep, I(R.si), I(R.di), E);

      // edge softmax per (dst, head)
      k_fill<<<g1((long)Nd * 8), 256, 0, stream>>>(smx, (long)Nd * 8, -INFINITY);
      k_segmax<<<g1((long)E * 8), 256, 0, stream>>>(smx, lgt, I(R.di), E);
      k_zero<<<g1((long)Nd * 8), 256, 0, stream>>>(ssm, (long)Nd * 8);
      k_expsum<<<g1((long)E * 8), 256, 0, stream>>>(lgt, smx, ssm, I(R.di), E);
      k_alpha<<<g1((long)E * 8), 256, 0, stream>>>(lgt, ssm, I(R.di), E);

      // message aggregation
      long nd_el = (long)Nd * D_MODEL;
      k_zero<<<g1(nd_el), 256, 0, stream>>>(gat, nd_el);
      k_agg<<<g1((long)E * (D_MODEL / 4)), 256, 0, stream>>>(gat, lgt, fs, I(R.si), I(R.di), E);

      if (R.has_e) {
        // GAT output: hn[dt] += gat + bias
        k_addbias<<<g1(nd_el), 256, 0, stream>>>(hn + HOFF[R.dt], gat, bias, nd_el);
      } else {
        // GATConvWrap: y = gat + fd + bias; z = y@W_out + b_out; BN; accumulate
        const float* W_out = F(LB + R.po + 1);
        const float* b_out = F(LB + R.po + 4);
        const float* beta  = F(LB + R.po + 5);
        const float* gamma = F(LB + R.po + 7);
        k_y<<<g1(nd_el), 256, 0, stream>>>(fs, gat, fdp, bias, nd_el);  // y into fs
        cvt(a16, fs, nd_el);
        cvtT(w16t, W_out, D_MODEL, D_MODEL);
        gemm(a16, w16t, b_out, gat, Nd, D_MODEL, D_MODEL, 1, 0);        // z into gat
        k_bnstats<<<D_MODEL, 256, 0, stream>>>(bmu, bvr, gat, Nd);
        k_bnapply<<<g1(nd_el), 256, 0, stream>>>(hn + HOFF[R.dt], gat, bmu, bvr, gamma, beta, nd_el);
      }
    }
    cur ^= 1;
  }

  // ---- classifier heads ----
  float* out = (float*)d_out;
  float* hfin = hbuf[cur];
  // doc: relu -> 768x512(relu) -> 512x56(relu) -> 56x2   (cls_g: leaves 35..40)
  k_relu_cvt<<<g1(1000L * D_MODEL / 4), 256, 0, stream>>>(a16, hfin + HOFF[0], 1000L * D_MODEL);
  cvtT(w16t, F(35), 768, 512);
  gemm(a16, w16t, F(36), fd, 1000, 512, 768, 1, 1);
  cvt(a16, fd, 1000L * 512);
  cvtT(w16t, F(37), 512, 56);
  gemm(a16, w16t, F(38), gat, 1000, 56, 512, 1, 1);
  cvt(a16, gat, 1000L * 56);
  cvtT(w16t, F(39), 56, 2);
  gemm(a16, w16t, F(40), out, 1000, 2, 56, 1, 0);
  k_i2f<<<g1(1000), 256, 0, stream>>>(out + 2000, I(27), 1000);
  // event: cls_e leaves 29..34
  k_relu_cvt<<<g1(8000L * D_MODEL / 4), 256, 0, stream>>>(a16, hfin + HOFF[2], 8000L * D_MODEL);
  cvtT(w16t, F(29), 768, 512);
  gemm(a16, w16t, F(30), fd, 8000, 512, 768, 1, 1);
  cvt(a16, fd, 8000L * 512);
  cvtT(w16t, F(31), 512, 56);
  gemm(a16, w16t, F(32), gat, 8000, 56, 512, 1, 1);
  cvt(a16, gat, 8000L * 56);
  cvtT(w16t, F(33), 56, 2);
  gemm(a16, w16t, F(34), out + 3000, 8000, 2, 56, 1, 0);
  k_i2f<<<g1(8000), 256, 0, stream>>>(out + 3000 + 16000, I(28), 8000);
}
